// HypergraphConvolution_13975823581873
// MI455X (gfx1250) — matrix operand order repaired and checked by reading, not compiled, not run on hardware
//
#include <hip/hip_runtime.h>

#define N_NODES  100000
#define N_HEDGES 20000
#define N_EDGES  1600000
#define DIM      160
#define LN_EPS   1e-5f

#define NT_TILES (DIM / 16)   // 10 N-tiles
#define KT_TILES (DIM / 32)   // 5  K-tiles
#define FRAG_ELEMS 512        // 32 lanes * 16 f16 per B-fragment

typedef __attribute__((ext_vector_type(16))) _Float16 v16h;
typedef __attribute__((ext_vector_type(8)))  float    v8f;

// ---------------------------------------------------------------- zero fill
__global__ void k_zero(float* __restrict__ p, long n) {
  long i = (long)blockIdx.x * blockDim.x + threadIdx.x;
  long stride = (long)gridDim.x * blockDim.x;
  for (; i < n; i += stride) p[i] = 0.0f;
}

// ---------------------------------------------- pack W (f32 row-major) into
// f16 B-fragments. Fragment (nt, kt): element (lane, i) = W[n][k] with
// n = nt*16 + (lane&15), k = kt*32 + (lane>>4)*8 + i (i<8), kt*32 + 16 + (lane>>4)*8 + (i-8) (i>=8). Lane data contiguous
// (32 B) so GEMM loads it with two global_load_b128, no conversions.
__global__ void k_pack_w(const float* __restrict__ W, _Float16* __restrict__ Wp) {
  int t = blockIdx.x * blockDim.x + threadIdx.x;
  if (t >= DIM * DIM) return;
  int tile = t / FRAG_ELEMS;        // 0..49
  int r    = t % FRAG_ELEMS;
  int lane = r / 16;
  int i    = r % 16;
  int nt = tile / KT_TILES;
  int kt = tile % KT_TILES;
  int n = nt * 16 + (lane & 15);
  int k = kt * 32 + (lane >> 4) * 8 + (i < 8 ? i : 16 + (i - 8));
  Wp[t] = (_Float16)W[(long)n * DIM + k];
}

// ------------------------------------------- gather + scatter-add (+count)
__global__ void k_scatter_mean_accum(const float* __restrict__ src,
                                     const int*   __restrict__ gather_idx,
                                     const int*   __restrict__ scatter_idx,
                                     float* __restrict__ dst_sum,
                                     float* __restrict__ dst_cnt) {
  const int CH = DIM / 4;  // 40 float4 chunks per row
  long idx = (long)blockIdx.x * blockDim.x + threadIdx.x;
  const long total = (long)N_EDGES * CH;
  if (idx >= total) return;
  int e = (int)(idx / CH);
  int c = (int)(idx % CH);
  int s = gather_idx[e];
  int d = scatter_idx[e];
  const float4 v = *(const float4*)(src + (long)s * DIM + c * 4);
  float* out = dst_sum + (long)d * DIM + c * 4;
  atomicAdd(out + 0, v.x);
  atomicAdd(out + 1, v.y);
  atomicAdd(out + 2, v.z);
  atomicAdd(out + 3, v.w);
  if (c == 0) atomicAdd(dst_cnt + d, 1.0f);
}

// ------------------------------------------------------------ WMMA helpers
// A-frag: 16-bit A 16x32 layout. Lane L: M = L&15. Elements 0..7 -> K = k0+ahi+i,
// elements 8..15 -> K = k0+16+ahi+i, with ahi = 8*(L>=16).
__device__ inline v16h load_a_frag(const _Float16* __restrict__ Arow, int k0, int ahi) {
  v16h a;
#pragma unroll
  for (int i = 0; i < 8; ++i) a[i] = Arow[k0 + ahi + i];
#pragma unroll
  for (int i = 0; i < 8; ++i) a[8 + i] = Arow[k0 + 16 + ahi + i];
  return a;
}

// B-frag from pre-packed buffer: lane's 16 f16 are contiguous (two b128 loads).
__device__ inline v16h load_b_frag_packed(const _Float16* __restrict__ Wp,
                                          int nt, int kt, int lane) {
  return *(const v16h*)(Wp + (long)(nt * KT_TILES + kt) * FRAG_ELEMS + lane * 16);
}

// ------------------------------ Stage 1: mean -> GEMM(W_he) -> +bias -> ReLU
// Two N-tiles in flight: independent accumulators fill WMMA hazard slots.
__global__ __launch_bounds__(32) void k_he_gemm_relu(
    const float* __restrict__ he_sum, const float* __restrict__ he_cnt,
    const _Float16* __restrict__ Wp, const float* __restrict__ bias,
    float* __restrict__ he_act) {
  __shared__ _Float16 A[16][DIM];
  __shared__ float    inv[16];
  const int lane = threadIdx.x;
  const int e0   = blockIdx.x * 16;

  if (lane < 16) inv[lane] = 1.0f / fmaxf(he_cnt[e0 + lane], 1.0f);
  __syncthreads();

#pragma unroll
  for (int i = 0; i < (16 * DIM) / 32; ++i) {
    int flat = i * 32 + lane;
    int r = flat / DIM, c = flat % DIM;
    A[r][c] = (_Float16)(he_sum[(long)(e0 + r) * DIM + c] * inv[r]);
  }
  __syncthreads();

  const int m    = lane & 15;
  const int half = lane >> 4;
  const int ahi  = half * 8;
  const int nl   = lane & 15;

  for (int np = 0; np < NT_TILES; np += 2) {
    v8f acc0 = {}, acc1 = {};
#pragma unroll
    for (int kt = 0; kt < KT_TILES; ++kt) {
      v16h a  = load_a_frag(&A[m][0], kt * 32, ahi);
      v16h b0 = load_b_frag_packed(Wp, np + 0, kt, lane);
      v16h b1 = load_b_frag_packed(Wp, np + 1, kt, lane);
      acc0 = __builtin_amdgcn_wmma_f32_16x16x32_f16(false, a, false, b0,
                                                    (short)0, acc0, false, false);
      acc1 = __builtin_amdgcn_wmma_f32_16x16x32_f16(false, a, false, b1,
                                                    (short)0, acc1, false, false);
    }
    float bv0 = bias[(np + 0) * 16 + nl];
    float bv1 = bias[(np + 1) * 16 + nl];
#pragma unroll
    for (int r = 0; r < 8; ++r) {
      int mo = r + half * 8;       // C layout: M = r (lanes 0-15) / r+8 (16-31)
      long rowoff = (long)(e0 + mo) * DIM;
      he_act[rowoff + (np + 0) * 16 + nl] = fmaxf(acc0[r] + bv0, 0.0f);
      he_act[rowoff + (np + 1) * 16 + nl] = fmaxf(acc1[r] + bv1, 0.0f);
    }
  }
}

// --- Stage 2: mean -> GEMM(W_node) -> +bias +residual -> LayerNorm -> ReLU
__global__ __launch_bounds__(32) void k_node_gemm_ln_relu(
    const float* __restrict__ node_sum, const float* __restrict__ node_cnt,
    const _Float16* __restrict__ Wp, const float* __restrict__ bias,
    const float* __restrict__ h_in,
    const float* __restrict__ gamma, const float* __restrict__ beta,
    float* __restrict__ out) {
  __shared__ _Float16 A[16][DIM];
  __shared__ float    OUT[16][DIM];
  __shared__ float    inv[16];
  __shared__ float    mu[16], rs[16];
  const int lane = threadIdx.x;
  const int v0   = blockIdx.x * 16;

  if (lane < 16) inv[lane] = 1.0f / fmaxf(node_cnt[v0 + lane], 1.0f);
  __syncthreads();

#pragma unroll
  for (int i = 0; i < (16 * DIM) / 32; ++i) {
    int flat = i * 32 + lane;
    int r = flat / DIM, c = flat % DIM;
    A[r][c] = (_Float16)(node_sum[(long)(v0 + r) * DIM + c] * inv[r]);
  }
  __syncthreads();

  const int m    = lane & 15;
  const int half = lane >> 4;
  const int ahi  = half * 8;
  const int nl   = lane & 15;

  for (int np = 0; np < NT_TILES; np += 2) {
    v8f acc0 = {}, acc1 = {};
#pragma unroll
    for (int kt = 0; kt < KT_TILES; ++kt) {
      v16h a  = load_a_frag(&A[m][0], kt * 32, ahi);
      v16h b0 = load_b_frag_packed(Wp, np + 0, kt, lane);
      v16h b1 = load_b_frag_packed(Wp, np + 1, kt, lane);
      acc0 = __builtin_amdgcn_wmma_f32_16x16x32_f16(false, a, false, b0,
                                                    (short)0, acc0, false, false);
      acc1 = __builtin_amdgcn_wmma_f32_16x16x32_f16(false, a, false, b1,
                                                    (short)0, acc1, false, false);
    }
    float bv0 = bias[(np + 0) * 16 + nl];
    float bv1 = bias[(np + 1) * 16 + nl];
#pragma unroll
    for (int r = 0; r < 8; ++r) {
      int mo = r + half * 8;
      long rowoff = (long)(v0 + mo) * DIM;
      OUT[mo][(np + 0) * 16 + nl] =
          acc0[r] + bv0 + h_in[rowoff + (np + 0) * 16 + nl];
      OUT[mo][(np + 1) * 16 + nl] =
          acc1[r] + bv1 + h_in[rowoff + (np + 1) * 16 + nl];
    }
  }
  __syncthreads();

  if (lane < 16) {
    float s = 0.0f, s2 = 0.0f;
#pragma unroll 4
    for (int j = 0; j < DIM; ++j) {
      float x = OUT[lane][j];
      s += x; s2 += x * x;
    }
    float m1  = s * (1.0f / DIM);
    float var = s2 * (1.0f / DIM) - m1 * m1;
    mu[lane] = m1;
    rs[lane] = rsqrtf(var + LN_EPS);
  }
  __syncthreads();

#pragma unroll
  for (int i = 0; i < (16 * DIM) / 32; ++i) {
    int flat = i * 32 + lane;
    int r = flat / DIM, c = flat % DIM;
    float v = (OUT[r][c] - mu[r]) * rs[r] * gamma[c] + beta[c];
    out[(long)(v0 + r) * DIM + c] = fmaxf(v, 0.0f);
  }
}

// --------------------------------------------------------------- dispatch
extern "C" void kernel_launch(void* const* d_in, const int* in_sizes, int n_in,
                              void* d_out, int out_size, void* d_ws, size_t ws_size,
                              hipStream_t stream) {
  const float* h           = (const float*)d_in[0];
  const int*   edge_nodes  = (const int*)d_in[1];
  const int*   edge_hedges = (const int*)d_in[2];
  // d_in[3] = num_hyperedges scalar (compile-time constant here)
  const float* W_he   = (const float*)d_in[4];
  const float* b_he   = (const float*)d_in[5];
  const float* W_node = (const float*)d_in[6];
  const float* b_node = (const float*)d_in[7];
  const float* gamma  = (const float*)d_in[8];
  const float* beta   = (const float*)d_in[9];

  float* ws       = (float*)d_ws;
  float* he_sum   = ws;                                  // E*D f32
  float* he_cnt   = he_sum + (long)N_HEDGES * DIM;       // E   f32
  float* he_act   = he_cnt + N_HEDGES;                   // E*D f32
  float* node_sum = he_act + (long)N_HEDGES * DIM;       // N*D f32
  float* node_cnt = node_sum + (long)N_NODES * DIM;      // N   f32
  _Float16* Wp_he   = (_Float16*)(node_cnt + N_NODES);   // D*D f16 (32B-aligned)
  _Float16* Wp_node = Wp_he + (long)DIM * DIM;           // D*D f16

  // one-time (per launch) weight packing: f32 -> f16 WMMA B-fragment layout
  k_pack_w<<<(DIM * DIM + 255) / 256, 256, 0, stream>>>(W_he, Wp_he);
  k_pack_w<<<(DIM * DIM + 255) / 256, 256, 0, stream>>>(W_node, Wp_node);

  const long z1 = (long)N_HEDGES * DIM + N_HEDGES;
  const long z2 = (long)N_NODES * DIM + N_NODES;
  k_zero<<<2048, 256, 0, stream>>>(he_sum, z1);
  k_zero<<<8192, 256, 0, stream>>>(node_sum, z2);

  const long total_work = (long)N_EDGES * (DIM / 4);
  const int  sblocks    = (int)((total_work + 255) / 256);

  // node -> hyperedge accumulation
  k_scatter_mean_accum<<<sblocks, 256, 0, stream>>>(h, edge_nodes, edge_hedges,
                                                    he_sum, he_cnt);
  // he = relu(mean @ W_he^T + b_he)
  k_he_gemm_relu<<<N_HEDGES / 16, 32, 0, stream>>>(he_sum, he_cnt, Wp_he, b_he,
                                                   he_act);
  // hyperedge -> node accumulation
  k_scatter_mean_accum<<<sblocks, 256, 0, stream>>>(he_act, edge_hedges, edge_nodes,
                                                    node_sum, node_cnt);
  // out = relu(LN(mean @ W_node^T + b_node + h))
  k_node_gemm_ln_relu<<<N_NODES / 16, 32, 0, stream>>>(node_sum, node_cnt, Wp_node,
                                                       b_node, h, gamma, beta,
                                                       (float*)d_out);
}
